// FlashQwenAttention_63213328662894
// MI455X (gfx1250) — matrix operand order repaired and checked by reading, not compiled, not run on hardware
//
#include <hip/hip_runtime.h>
#include <hip/hip_bf16.h>

typedef __bf16 bf16;
typedef __attribute__((ext_vector_type(4)))  float    v4f;
typedef __attribute__((ext_vector_type(8)))  float    v8f;
typedef __attribute__((ext_vector_type(8)))  __bf16   v8bf;
typedef __attribute__((ext_vector_type(16))) __bf16   v16bf;
typedef __attribute__((ext_vector_type(4)))  unsigned u32x4;
typedef __attribute__((ext_vector_type(4)))  int      i32x4;
typedef __attribute__((ext_vector_type(8)))  int      i32x8;

constexpr int kH = 16, kD = 128, kHID = 2048, kB = 2, kS = 2048, kT = 4096, kN3 = 6144;

#if __has_builtin(__builtin_amdgcn_tensor_load_to_lds) && __has_builtin(__builtin_amdgcn_s_wait_tensorcnt)
#define HAVE_TDM 1
#else
#define HAVE_TDM 0
#endif

__device__ __forceinline__ v16bf cat8(v8bf lo, v8bf hi) {
  return __builtin_shufflevector(lo, hi, 0,1,2,3,4,5,6,7,8,9,10,11,12,13,14,15);
}
__device__ __forceinline__ v8f wmma_bf16(v16bf a, v16bf b, v8f c) {
  return __builtin_amdgcn_wmma_f32_16x16x32_bf16(false, a, false, b, (short)0, c, false, false);
}
// A-operand fragment: lane holds row; 16 contiguous K values starting at 2*lh8
__device__ __forceinline__ v16bf afrag_ld(const bf16* row, int lh8) {
  return cat8(*(const v8bf*)(row + 2 * lh8), *(const v8bf*)(row + 2 * lh8 + 8));
}
// B-operand fragment: lane holds column; 16 contiguous K values starting at lh16
__device__ __forceinline__ v16bf bfrag_ld(const bf16* col, int lh16) {
  return cat8(*(const v8bf*)(col + lh16), *(const v8bf*)(col + lh16 + 8));
}

#if HAVE_TDM
// Issue a 2D TDM tensor_load_to_lds. All units are 4-byte dwords (data_size=2).
// pad_interval_code: 0=2,1=4,2=8,3=16,4=32,5=64 dwords; pad_amount_code: n -> n+1 dwords.
__device__ __forceinline__ void tdm_load_2d(unsigned lds_off, const void* gptr,
                                            int tensor_d0, int tensor_d1,
                                            int tile_d0, int tile_d1,
                                            long long stride0_dw,
                                            int pad_interval_code, int pad_amount_code) {
  const unsigned long long ga = (unsigned long long)(uintptr_t)gptr;
  u32x4 g0;
  g0[0] = 1u;                                              // count=1, user-mode
  g0[1] = lds_off;                                         // lds_addr (bytes)
  g0[2] = (unsigned)(ga & 0xFFFFFFFFu);                    // global_addr[31:0]
  g0[3] = (unsigned)((ga >> 32) & 0x01FFFFFFu) | 0x80000000u;  // addr[56:32] | type=2
  i32x8 g1;
  g1[0] = (2 << 16) | (1 << 20) | (pad_interval_code << 22) | (pad_amount_code << 25);
  g1[1] = (tensor_d0 & 0xFFFF) << 16;                      // tensor_dim0[15:0]
  g1[2] = ((tensor_d0 >> 16) & 0xFFFF) | ((tensor_d1 & 0xFFFF) << 16);
  g1[3] = ((tensor_d1 >> 16) & 0xFFFF) | ((tile_d0 & 0xFFFF) << 16);
  g1[4] = (tile_d1 & 0xFFFF);                              // tile_dim1, tile_dim2=0
  g1[5] = (int)(stride0_dw & 0xFFFFFFFFll);                // tensor_dim0_stride[31:0]
  g1[6] = (int)((stride0_dw >> 32) & 0xFFFFll);            // stride[47:32]; dim1_stride=0
  g1[7] = 0;
  const i32x4 z4 = {0, 0, 0, 0};
#if __has_include(<hip/amd_detail/amd_gfx1250_TDM.h>)
  const i32x8 z8 = {0, 0, 0, 0, 0, 0, 0, 0};
  __builtin_amdgcn_tensor_load_to_lds(g0, g1, z4, z4, z8, 0);
#else
  __builtin_amdgcn_tensor_load_to_lds(g0, g1, z4, z4, 0);
#endif
}
#endif

// ---------------------------------------------------------------------------
// Generic bf16-WMMA GEMM: C[M,N] = A[M,K] * W[K,N] (+bias). A fp32 or bf16.
// Block tile 128x64, 8 waves each doing a 32x32 tile, K step 32.
// bf16-A tiles are staged into LDS by the Tensor Data Mover.
// ---------------------------------------------------------------------------
template<bool A_IS_BF16>
__global__ void __launch_bounds__(256)
gemm_wmma_bf16(const void* __restrict__ Ap, const float* __restrict__ Wp,
               const float* __restrict__ bias, float* __restrict__ Cp,
               int M, int N, int K) {
  constexpr int BM = 128, BN = 64, BK = 32, LP = BK + 8;
  __shared__ bf16 As[BM][LP];   // row-major (A-operand), pitch 80 B
  __shared__ bf16 Bs[BN][LP];   // transposed: Bs[n][k] (B-operand)
  const int tid  = threadIdx.x;
  const int wave = tid >> 5, lane = tid & 31;
  const int lrow = lane & 15;
  const int lh8  = (lane >> 4) * 8;
  const int lh16 = (lane >> 4) * 16;
  const int m0 = blockIdx.x * BM;
  const int n0 = blockIdx.y * BN;
  const int wm = (wave & 3) * 32;
  const int wn = (wave >> 2) * 32;
#if HAVE_TDM
  const unsigned as_lds = (unsigned)(uintptr_t)&As[0][0];
#endif

  const v8f vzero = {};
  v8f acc[2][2];
#pragma unroll
  for (int i = 0; i < 2; ++i)
#pragma unroll
    for (int j = 0; j < 2; ++j) acc[i][j] = vzero;

  const int ar = tid >> 1, ac = (tid & 1) * 16;   // A tile: 128 rows x 32 cols
  const int bkr = tid >> 3, bc = (tid & 7) * 8;   // W tile: 32 rows x 64 cols

  for (int k0 = 0; k0 < K; k0 += BK) {
    __syncthreads();
    if constexpr (A_IS_BF16) {
#if HAVE_TDM
      if (wave == 0)
        tdm_load_2d(as_lds, (const bf16*)Ap + (size_t)m0 * K + k0,
                    BK / 2, BM, BK / 2, BM, (long long)(K / 2),
                    /*interval 16dw*/ 3, /*pad 4dw*/ 3);
#else
      const bf16* src = (const bf16*)Ap + (size_t)(m0 + ar) * K + (k0 + ac);
      *(v8bf*)&As[ar][ac]     = *(const v8bf*)src;
      *(v8bf*)&As[ar][ac + 8] = *(const v8bf*)(src + 8);
      if (k0 + BK < K) __builtin_prefetch(src + BK, 0, 0);
#endif
    } else {
      const float* src = (const float*)Ap + (size_t)(m0 + ar) * K + (k0 + ac);
      v4f f0 = *(const v4f*)src;
      v4f f1 = *(const v4f*)(src + 4);
      v4f f2 = *(const v4f*)(src + 8);
      v4f f3 = *(const v4f*)(src + 12);
      v8bf x0, x1;
#pragma unroll
      for (int i = 0; i < 4; ++i) {
        x0[i] = (bf16)f0[i]; x0[i + 4] = (bf16)f1[i];
        x1[i] = (bf16)f2[i]; x1[i + 4] = (bf16)f3[i];
      }
      *(v8bf*)&As[ar][ac]     = x0;
      *(v8bf*)&As[ar][ac + 8] = x1;
      if (k0 + BK < K) __builtin_prefetch(src + BK, 0, 0);
    }
    {
      const float* src = Wp + (size_t)(k0 + bkr) * N + (n0 + bc);
      v4f f0 = *(const v4f*)src;
      v4f f1 = *(const v4f*)(src + 4);
#pragma unroll
      for (int i = 0; i < 4; ++i) {
        Bs[bc + i][bkr]     = (bf16)f0[i];
        Bs[bc + 4 + i][bkr] = (bf16)f1[i];
      }
      if (k0 + BK < K) __builtin_prefetch(src + (size_t)BK * N, 0, 0);
    }
#if HAVE_TDM
    if (A_IS_BF16 && wave == 0) __builtin_amdgcn_s_wait_tensorcnt(0);
#endif
    __syncthreads();
#pragma unroll
    for (int sm = 0; sm < 2; ++sm) {
      v16bf a = afrag_ld(&As[wm + sm * 16 + lrow][0], lh8);
#pragma unroll
      for (int sn = 0; sn < 2; ++sn) {
        v16bf b = bfrag_ld(&Bs[wn + sn * 16 + lrow][0], lh16);
        acc[sm][sn] = wmma_bf16(a, b, acc[sm][sn]);
      }
    }
  }
#pragma unroll
  for (int sm = 0; sm < 2; ++sm)
#pragma unroll
    for (int sn = 0; sn < 2; ++sn) {
      const int row0 = m0 + wm + sm * 16 + (lane >> 4) * 8;
      const int col  = n0 + wn + sn * 16 + lrow;
      const float bv = bias ? bias[col] : 0.0f;
#pragma unroll
      for (int r = 0; r < 8; ++r)
        Cp[(size_t)(row0 + r) * N + col] = acc[sm][sn][r] + bv;
    }
}

// ---------------------------------------------------------------------------
// RoPE + split: qkv fp32 -> q/k/v bf16 for attention, rope'd k & raw v fp32
// straight into the kc/vc outputs (slots == arange -> identity scatter).
// ---------------------------------------------------------------------------
__global__ void __launch_bounds__(128)
rope_split(const float* __restrict__ qkv, const float* __restrict__ cs,
           const float* __restrict__ sn, bf16* __restrict__ qb,
           bf16* __restrict__ kb, bf16* __restrict__ vb,
           float* __restrict__ kc, float* __restrict__ vc) {
  const int th = blockIdx.x;
  const int t = th >> 4, h = th & 15;
  const int d = threadIdx.x;
  const float* base = qkv + (size_t)t * kN3;
  const float q = base[h * kD + d];
  const float k = base[kHID + h * kD + d];
  const float v = base[2 * kHID + h * kD + d];
  const int   dp = (d < 64) ? d + 64 : d - 64;
  const float qp = base[h * kD + dp];
  const float kp = base[kHID + h * kD + dp];
  const int   di = (d < 64) ? d : d - 64;
  const float c = cs[t * 64 + di];
  const float s = sn[t * 64 + di];
  const float qr = (d < 64) ? (q * c - qp * s) : (q * c + qp * s);
  const float kr = (d < 64) ? (k * c - kp * s) : (k * c + kp * s);
  const size_t o = (size_t)th * kD + d;
  qb[o] = (bf16)qr;
  kb[o] = (bf16)kr;
  vb[o] = (bf16)v;
  kc[o] = kr;
  vc[o] = v;
}

// ---------------------------------------------------------------------------
// Causal flash attention. Block = (128 q rows, head, batch); 8 waves each own
// 16 q rows. K tiles DMA'd to LDS by the TDM; V tiles cooperatively
// transposed; P staged via per-wave LDS (C-layout -> A-operand layout).
// ---------------------------------------------------------------------------
__global__ void __launch_bounds__(256)
flash_attn_wmma(const bf16* __restrict__ qb, const bf16* __restrict__ kb,
                const bf16* __restrict__ vb, bf16* __restrict__ attnb) {
  constexpr int QT = 128, KT = 32;
  __shared__ bf16 Ks[KT][kD + 8];      // K tile, row-major over d, pitch 272 B
  __shared__ bf16 Vt[kD][KT + 8];      // V tile transposed: Vt[d][kv]
  __shared__ bf16 Ps[8][16][KT + 8];   // per-wave P staging
  const int qt = blockIdx.x, h = blockIdx.y, b = blockIdx.z;
  const int tid = threadIdx.x, wave = tid >> 5, lane = tid & 31;
  const int lrow = lane & 15, lh8 = (lane >> 4) * 8, lh16 = (lane >> 4) * 16;
  const int q0 = qt * QT + wave * 16;
#if HAVE_TDM
  const unsigned ks_lds = (unsigned)(uintptr_t)&Ks[0][0];
#endif

  v16bf qfrag[4];
  {
    const bf16* qrow = qb + ((size_t)(b * kS + q0 + lrow) * kH + h) * kD;
#pragma unroll
    for (int c = 0; c < 4; ++c) qfrag[c] = afrag_ld(qrow + c * 32, lh8);
  }
  const v8f vzero = {};
  v8f acc[8];
  float mrow[8], lsum[8];
#pragma unroll
  for (int r = 0; r < 8; ++r) { acc[r] = vzero; mrow[r] = -1e30f; lsum[r] = 0.0f; }

  const float scale = 0.08838834764831845f;  // 1/sqrt(128)
  const int kvend = qt * QT + QT;
  const int ldr = tid >> 3, ldc = (tid & 7) * 16;

  for (int j = 0; j < kvend; j += KT) {
    __syncthreads();
#if HAVE_TDM
    if (wave == 0)
      tdm_load_2d(ks_lds, kb + ((size_t)(b * kS + j) * kH + h) * kD,
                  kD / 2, KT, kD / 2, KT, (long long)(kH * kD / 2),
                  /*interval 64dw*/ 5, /*pad 4dw*/ 3);
#else
    {
      const bf16* ksrc = kb + ((size_t)(b * kS + j + ldr) * kH + h) * kD + ldc;
      *(v8bf*)&Ks[ldr][ldc]     = *(const v8bf*)ksrc;
      *(v8bf*)&Ks[ldr][ldc + 8] = *(const v8bf*)(ksrc + 8);
    }
#endif
    {
      const bf16* vsrc = vb + ((size_t)(b * kS + j + ldr) * kH + h) * kD + ldc;
      v8bf x0 = *(const v8bf*)vsrc;
      v8bf x1 = *(const v8bf*)(vsrc + 8);
#pragma unroll
      for (int i = 0; i < 8; ++i) {
        Vt[ldc + i][ldr]     = x0[i];
        Vt[ldc + 8 + i][ldr] = x1[i];
      }
    }
#if HAVE_TDM
    if (wave == 0) __builtin_amdgcn_s_wait_tensorcnt(0);
#endif
    __syncthreads();

    // scores: two 16x16 tiles covering 32 kv columns
    v8f s[2];
#pragma unroll
    for (int jt = 0; jt < 2; ++jt) {
      v8f sc = vzero;
#pragma unroll
      for (int c = 0; c < 4; ++c) {
        v16bf kf = bfrag_ld(&Ks[jt * 16 + lrow][c * 32], lh16);
        sc = wmma_bf16(qfrag[c], kf, sc);
      }
      s[jt] = sc;
    }
    // scale + causal mask
#pragma unroll
    for (int jt = 0; jt < 2; ++jt)
#pragma unroll
      for (int r = 0; r < 8; ++r) {
        const int qpos = q0 + r + (lane >> 4) * 8;
        const int kpos = j + jt * 16 + lrow;
        const float v = s[jt][r] * scale;
        s[jt][r] = (kpos <= qpos) ? v : -1e30f;
      }
    // online softmax (row stats live redundantly in each 16-lane group)
#pragma unroll
    for (int r = 0; r < 8; ++r) {
      float mx = fmaxf(s[0][r], s[1][r]);
      mx = fmaxf(mx, __shfl_xor(mx, 1, 32));
      mx = fmaxf(mx, __shfl_xor(mx, 2, 32));
      mx = fmaxf(mx, __shfl_xor(mx, 4, 32));
      mx = fmaxf(mx, __shfl_xor(mx, 8, 32));
      const float mnew  = fmaxf(mrow[r], mx);
      const float alpha = __expf(mrow[r] - mnew);
      mrow[r] = mnew;
      const float e0 = __expf(s[0][r] - mnew);
      const float e1 = __expf(s[1][r] - mnew);
      s[0][r] = e0; s[1][r] = e1;
      float rs = e0 + e1;
      rs += __shfl_xor(rs, 1, 32);
      rs += __shfl_xor(rs, 2, 32);
      rs += __shfl_xor(rs, 4, 32);
      rs += __shfl_xor(rs, 8, 32);
      lsum[r] = lsum[r] * alpha + rs;
#pragma unroll
      for (int nt = 0; nt < 8; ++nt) acc[nt][r] *= alpha;
    }
    // C-layout -> A-layout for P via per-wave LDS
#pragma unroll
    for (int jt = 0; jt < 2; ++jt)
#pragma unroll
      for (int r = 0; r < 8; ++r)
        Ps[wave][r + (lane >> 4) * 8][jt * 16 + lrow] = (bf16)s[jt][r];
    asm volatile("s_wait_dscnt 0" ::: "memory");
    v16bf pf = afrag_ld(&Ps[wave][lrow][0], lh8);
#pragma unroll
    for (int nt = 0; nt < 8; ++nt) {
      v16bf vf = bfrag_ld(&Vt[nt * 16 + lrow][0], lh16);
      acc[nt] = wmma_bf16(pf, vf, acc[nt]);
    }
  }
  // finalize: divide by row sum, emit bf16 attn output (T, H*D)
#pragma unroll
  for (int nt = 0; nt < 8; ++nt)
#pragma unroll
    for (int r = 0; r < 8; ++r) {
      const int q = q0 + r + (lane >> 4) * 8;
      const float o = acc[nt][r] / lsum[r];
      attnb[(size_t)(b * kS + q) * kHID + h * kD + nt * 16 + lrow] = (bf16)o;
    }
}

// ---------------------------------------------------------------------------
extern "C" void kernel_launch(void* const* d_in, const int* in_sizes, int n_in,
                              void* d_out, int out_size, void* d_ws, size_t ws_size,
                              hipStream_t stream) {
  (void)in_sizes; (void)n_in; (void)out_size; (void)ws_size;
  const float* hs   = (const float*)d_in[0];
  const float* cs   = (const float*)d_in[1];
  const float* sn   = (const float*)d_in[2];
  const float* wqkv = (const float*)d_in[3];
  const float* bqkv = (const float*)d_in[4];
  const float* wo   = (const float*)d_in[5];

  float* out = (float*)d_out;
  float* kc  = out + (size_t)kT * kHID;   // kc region (NB*BS*H*D == T*H*D)
  float* vc  = kc  + (size_t)kT * kHID;

  float* qkv  = (float*)d_ws;                       // T x 3*HID fp32
  bf16* qb    = (bf16*)(qkv + (size_t)kT * kN3);    // T x H x D bf16
  bf16* kbuf  = qb   + (size_t)kT * kHID;
  bf16* vbuf  = kbuf + (size_t)kT * kHID;
  bf16* attnb = vbuf + (size_t)kT * kHID;           // T x HID bf16

  gemm_wmma_bf16<false><<<dim3(kT / 128, kN3 / 64), dim3(256), 0, stream>>>(
      hs, wqkv, bqkv, qkv, kT, kN3, kHID);
  rope_split<<<dim3(kT * kH), dim3(128), 0, stream>>>(
      qkv, cs, sn, qb, kbuf, vbuf, kc, vc);
  flash_attn_wmma<<<dim3(kS / 128, kH, kB), dim3(256), 0, stream>>>(
      qb, kbuf, vbuf, attnb);
  gemm_wmma_bf16<true><<<dim3(kT / 128, kHID / 64), dim3(256), 0, stream>>>(
      attnb, wo, nullptr, out, kT, kHID, kHID);
}
